// MambaBlock_66005057405550
// MI455X (gfx1250) — matrix operand order repaired and checked by reading, not compiled, not run on hardware
//
#include <hip/hip_runtime.h>

#define BATCH      512
#define INPUT_DIM  32760
#define D_MODEL    16
#define D_INNER    32
#define D_STATE    16
#define D_CONV     4
#define L_SEQ      2048          // 32768 / 16 (padded)
#define CHUNK      64
#define NCHUNK     (L_SEQ / CHUNK)
#define NTHREADS   512
#define LN_EPS     1e-5f

typedef __attribute__((ext_vector_type(16))) _Float16 v16h;
typedef __attribute__((ext_vector_type(8)))  float    v8f;

__device__ __forceinline__ float fast_rcp(float x) { return __builtin_amdgcn_rcpf(x); }
__device__ __forceinline__ float silu(float x)     { return x * fast_rcp(1.f + __expf(-x)); }

// One 16x16 output tile of A(16x32,f16) x B(32x16,f16) with f32 accumulate.
// Layouts per CDNA5 ISA 7.12.2 (wave32):
//   A f16 16x32: lane m=lane&15; halves j: K = ((j&8)<<1) + ((lane>>4)&1)*8 + (j&7)
//   B f16 32x16: lane n=lane&15; halves j: K = ((j&8)<<1) + ((lane>>4)&1)*8 + (j&7)
__device__ __forceinline__ v8f wmma_tile_f16(const float* __restrict__ A, int lda,
                                             const float* __restrict__ B, int ldb,
                                             int lane) {
  v16h av, bv;
  const int m  = lane & 15;
  const int hi = (lane >> 4) & 1;
#pragma unroll
  for (int j = 0; j < 16; ++j) {
    const int k = ((j & 8) << 1) + hi * 8 + (j & 7);
    av[j] = (_Float16)A[m * lda + k];
  }
#pragma unroll
  for (int j = 0; j < 16; ++j) {
    const int k = ((j & 8) << 1) + hi * 8 + (j & 7);
    bv[j] = (_Float16)B[k * ldb + m];
  }
  v8f acc = {};
  return __builtin_amdgcn_wmma_f32_16x16x32_f16(false, av, false, bv,
                                                (short)0, acc, false, false);
}

__global__ __launch_bounds__(NTHREADS)
void mamba_fused_kernel(const float* __restrict__ x,
                        const float* __restrict__ in_w,     // (64,16)
                        const float* __restrict__ conv_w,   // (32,4)
                        const float* __restrict__ conv_b,   // (32)
                        const float* __restrict__ xp_w,     // (33,32)
                        const float* __restrict__ dt_w,     // (32,1)
                        const float* __restrict__ dt_b,     // (32)
                        const float* __restrict__ A_log,    // (32,16)
                        const float* __restrict__ Dp,       // (32)
                        const float* __restrict__ out_w,    // (16,32)
                        const float* __restrict__ gamma,
                        const float* __restrict__ beta,
                        float* __restrict__ out) {
  // ---- LDS (~69.4 KB) ----
  __shared__ float s_xz[CHUNK * 64];     // xc cols 0..31, z cols 32..63
  __shared__ float s_u[CHUNK * 32];      // silu(conv) -> later gated y (in place)
  __shared__ float s_ys[CHUNK * 32];     // x-tile staging (K-padded), later raw scan y
  __shared__ float s_xdbl[CHUNK * 48];   // dt | B(16) | C(16) | zero-pad
  __shared__ float s_delta[CHUNK * 32];  // softplus(delta); reused for LN reduction
  __shared__ float s_win[32 * 64];       // in_proj  weights, K-major, K padded to 32
  __shared__ float s_wxp[32 * 48];       // x_proj   weights, K-major, N padded to 48
  __shared__ float s_wout[32 * 16];      // out_proj weights, K-major
  __shared__ float s_carry[3 * 32];      // conv tail of previous chunk
  __shared__ float s_convw[32 * 4], s_convb[32], s_dtw[32], s_dtb[32], s_D[32];

  const int tid  = threadIdx.x;
  const int lane = tid & 31;
  const int wave = tid >> 5;
  const int b    = blockIdx.x;

  const float* __restrict__ xrow = x   + (size_t)b * INPUT_DIM;
  float*       __restrict__ orow = out + (size_t)b * INPUT_DIM;

  // ---- stage weights into LDS (K-major, zero padded) ----
  for (int i = tid; i < 32 * 64; i += NTHREADS) {
    const int k = i >> 6, e = i & 63;
    s_win[i] = (k < 16) ? in_w[e * 16 + k] : 0.f;
  }
  for (int i = tid; i < 32 * 48; i += NTHREADS) {
    const int k = i / 48, e = i % 48;
    s_wxp[i] = (e < 33) ? xp_w[e * 32 + k] : 0.f;
  }
  if (tid < 32 * 16) {
    const int k = tid >> 4, d = tid & 15;
    s_wout[tid] = out_w[d * 32 + k];
  }
  if (tid < 128) s_convw[tid] = conv_w[tid];
  if (tid < 32) {
    s_convb[tid] = conv_b[tid];
    s_dtw[tid]   = dt_w[tid];
    s_dtb[tid]   = dt_b[tid];
    s_D[tid]     = Dp[tid];
  }
  if (tid < 96) s_carry[tid] = 0.f;

  // scan-thread mapping: thread = channel*16 + state  (A_log flat index == tid)
  const int   sc   = tid >> 4;
  const int   ss   = tid & 15;
  const float Areg = -__expf(A_log[tid]);
  float hstate = 0.f;
  float lsum = 0.f, lsq = 0.f;
  __syncthreads();

  for (int ch = 0; ch < NCHUNK; ++ch) {
    const int cbase = ch * CHUNK;

    // ---- stage 1: load x tile vectorized (cols 0..15 real, 16..31 zero K-pad) ----
    if (tid < 256) {
      const int t = tid >> 2, g = tid & 3;            // 64 tokens x 4 float4
      const int idx = (cbase + t) * 16 + g * 4;       // multiple of 4; DIM%4==0
      float4 v = make_float4(0.f, 0.f, 0.f, 0.f);
      if (idx < INPUT_DIM) v = *(const float4*)(xrow + idx);
      *(float4*)&s_ys[t * 32 + g * 4] = v;
    } else {
      const int r = tid - 256;
      const int t = r >> 2, g = r & 3;
      *(float4*)&s_ys[t * 32 + 16 + g * 4] = make_float4(0.f, 0.f, 0.f, 0.f);
    }
    if (ch + 1 < NCHUNK && lane == 0)
      __builtin_prefetch(xrow + (size_t)(cbase + CHUNK) * 16 + wave * 256, 0, 3);
    __syncthreads();

    // ---- stage 2: in_proj  (4x4 tiles -> all 16 waves) ----
    {
      const int tm = wave >> 2, tn = wave & 3;
      v8f c = wmma_tile_f16(&s_ys[(tm * 16) * 32], 32, &s_win[tn * 16], 64, lane);
      const int n = lane & 15, hi = (lane >> 4) & 1;
#pragma unroll
      for (int i = 0; i < 8; ++i)
        s_xz[(tm * 16 + i + hi * 8) * 64 + tn * 16 + n] = c[i];
    }
    __syncthreads();

    // ---- stage 3: causal depthwise conv + SiLU -> s_u ----
#pragma unroll
    for (int i = tid; i < CHUNK * 32; i += NTHREADS) {
      const int t = i >> 5, cc = i & 31;
      float acc = s_convb[cc];
#pragma unroll
      for (int k = 0; k < D_CONV; ++k) {
        const int tt = t - 3 + k;
        const float xv = (tt >= 0) ? s_xz[tt * 64 + cc] : s_carry[(tt + 3) * 32 + cc];
        acc += xv * s_convw[cc * 4 + k];
      }
      s_u[i] = silu(acc);
    }
    __syncthreads();

    // ---- stage 4: x_proj  (4x3 tiles -> waves 0..11) ----
    if (wave < 12) {
      const int tm = wave >> 2;          // wave/3? no: 12 waves as 4x3: tm=wave/3
      const int tmm = wave / 3, tn = wave % 3;
      (void)tm;
      v8f c = wmma_tile_f16(&s_u[(tmm * 16) * 32], 32, &s_wxp[tn * 16], 48, lane);
      const int n = lane & 15, hi = (lane >> 4) & 1;
#pragma unroll
      for (int i = 0; i < 8; ++i)
        s_xdbl[(tmm * 16 + i + hi * 8) * 48 + tn * 16 + n] = c[i];
    }
    __syncthreads();

    // ---- stage 5: delta = softplus(dt*dt_w + dt_b); also save conv tail ----
#pragma unroll
    for (int i = tid; i < CHUNK * 32; i += NTHREADS) {
      const int t = i >> 5, cc = i & 31;
      const float d = s_xdbl[t * 48] * s_dtw[cc] + s_dtb[cc];
      s_delta[i] = (d > 20.f) ? d : __logf(1.f + __expf(d));
    }
    if (tid < 96) {                      // carry: readers (stage 3) are long done
      const int j = tid >> 5, cc = tid & 31;
      s_carry[tid] = s_xz[(CHUNK - 3 + j) * 64 + cc];
    }
    __syncthreads();

    // ---- stage 6: selective scan (thread = channel*16 + state) ----
#pragma unroll 4
    for (int t = 0; t < CHUNK; ++t) {
      const float dl = s_delta[t * 32 + sc];
      const float ut = s_u[t * 32 + sc];
      const float bt = s_xdbl[t * 48 + 1 + ss];
      const float ct = s_xdbl[t * 48 + 17 + ss];
      const float a  = __expf(dl * Areg);
      hstate = a * hstate + (dl * bt) * ut;
      float p = hstate * ct;
      p += __shfl_xor(p, 1);
      p += __shfl_xor(p, 2);
      p += __shfl_xor(p, 4);
      p += __shfl_xor(p, 8);
      if (ss == 0) s_ys[t * 32 + sc] = p;   // raw scan output
    }
    __syncthreads();

    // ---- stage 7: y = (ys + u*D) * silu(z), in place into s_u ----
#pragma unroll
    for (int i = tid; i < CHUNK * 32; i += NTHREADS) {
      const int t = i >> 5, cc = i & 31;
      const float z = s_xz[t * 64 + 32 + cc];
      s_u[i] = (s_ys[i] + s_u[i] * s_D[cc]) * silu(z);
    }
    __syncthreads();

    // ---- stage 8: out_proj (4 tiles -> waves 0..3) + residual + LN stats ----
    // no trailing barrier: waves 4..15 proceed to next chunk's x-tile load,
    // which touches only s_ys (whose last readers finished at the stage-7 barrier).
    if (wave < 4) {
      const int tm = wave;
      v8f c = wmma_tile_f16(&s_u[(tm * 16) * 32], 32, s_wout, 16, lane);
      const int n = lane & 15, hi = (lane >> 4) & 1;
#pragma unroll
      for (int i = 0; i < 8; ++i) {
        const int row = tm * 16 + i + hi * 8;
        const int idx = (cbase + row) * 16 + n;
        if (idx < INPUT_DIM) {
          const float hv = c[i] + xrow[idx];
          orow[idx] = hv;
          lsum += hv;
          lsq  += hv * hv;
        }
      }
    }
  }
  __syncthreads();

  // ---- deterministic block reduction of LN statistics (reuse s_delta) ----
  float* red = s_delta;                 // 2048 floats available
  red[tid]       = lsum;
  red[512 + tid] = lsq;
  __syncthreads();
  for (int off = 256; off > 0; off >>= 1) {
    if (tid < off) {
      red[tid]       += red[tid + off];
      red[512 + tid] += red[512 + tid + off];
    }
    __syncthreads();
  }
  const float mu  = red[0] * (1.f / INPUT_DIM);
  const float var = red[512] * (1.f / INPUT_DIM) - mu * mu;
  const float rs  = rsqrtf(var + LN_EPS);

  // ---- normalize in place ----
  for (int idx = tid; idx < INPUT_DIM; idx += NTHREADS) {
    const float hv = orow[idx];
    orow[idx] = (hv - mu) * rs * gamma[idx] + beta[idx];
  }
}

extern "C" void kernel_launch(void* const* d_in, const int* in_sizes, int n_in,
                              void* d_out, int out_size, void* d_ws, size_t ws_size,
                              hipStream_t stream) {
  (void)in_sizes; (void)n_in; (void)out_size; (void)d_ws; (void)ws_size;
  const float* x      = (const float*)d_in[0];
  const float* in_w   = (const float*)d_in[1];
  const float* conv_w = (const float*)d_in[2];
  const float* conv_b = (const float*)d_in[3];
  const float* xp_w   = (const float*)d_in[4];
  const float* dt_w   = (const float*)d_in[5];
  const float* dt_b   = (const float*)d_in[6];
  const float* A_log  = (const float*)d_in[7];
  const float* Dp     = (const float*)d_in[8];
  const float* out_w  = (const float*)d_in[9];
  const float* gamma  = (const float*)d_in[10];
  const float* beta   = (const float*)d_in[11];
  float* out = (float*)d_out;

  mamba_fused_kernel<<<dim3(BATCH), dim3(NTHREADS), 0, stream>>>(
      x, in_w, conv_w, conv_b, xp_w, dt_w, dt_b, A_log, Dp, out_w, gamma, beta, out);
}
